// CoattnConv_58162447123069
// MI455X (gfx1250) — compile-verified, matrix operand order checked by reading, not run on hardware
//
#include <hip/hip_runtime.h>
#include <hip/hip_bf16.h>
#include <math.h>

// ---------------------------------------------------------------------------
// CDNA5 (gfx1250) WMMA types
// ---------------------------------------------------------------------------
typedef __attribute__((ext_vector_type(16))) __bf16 v16bf;
typedef __attribute__((ext_vector_type(8)))  float  v8f;

#define WMMA_BF16(A, B, C) \
  __builtin_amdgcn_wmma_f32_16x16x32_bf16(false, (A), false, (B), (short)0, (C), false, false)

// gfx1250 hardware tanh if the toolchain exposes it, else ocml
#if __has_builtin(__builtin_amdgcn_tanhf)
#define TANHF(x) __builtin_amdgcn_tanhf(x)
#else
#define TANHF(x) tanhf(x)
#endif

#define NN 50000
#define EE 800000
#define HD 64

// ---------------------------------------------------------------------------
// ISA A-operand layout helper (16-bit A 16x32):
//   lane L (L<16): M=L,   K in {0..7}  (VGPR0-3) and {16..23} (VGPR4-7)
//   lane L+16    : M=L,   K in {8..15}           and {24..31}
// i.e. per lane: two contiguous 8-element K runs.
// ---------------------------------------------------------------------------
__device__ __forceinline__ int kofs(int lane, int j) {
  int v = j >> 1, p = j & 1;
  int hi = (lane >> 4) & 1;
  return (v < 4) ? ((hi ? 8 : 0) + 2 * v + p) : ((hi ? 24 : 16) + 2 * (v - 4) + p);
}

__device__ __forceinline__ unsigned fenc(float f) {  // order-preserving uint key
  unsigned u = __float_as_uint(f);
  return (u & 0x80000000u) ? ~u : (u | 0x80000000u);
}
__device__ __forceinline__ float fdec(unsigned u) {
  unsigned v = (u & 0x80000000u) ? (u & 0x7FFFFFFFu) : ~u;
  return __uint_as_float(v);
}

// Load an A fragment (16x32 K-chunk kc of a 16x64 fp32 tile) straight from a
// row pointer; lanes l and l+16 read the same row, different K runs.
__device__ __forceinline__ v16bf load_a_frag(const float* rowp, int lane, int kc) {
  int b0 = (((lane >> 4) & 1) ? 8 : 0) + 32 * kc;
  v16bf a;
#pragma unroll
  for (int j = 0; j < 8; ++j) a[j] = (__bf16)rowp[b0 + j];
#pragma unroll
  for (int j = 0; j < 8; ++j) a[8 + j] = (__bf16)rowp[b0 + 16 + j];
  return a;
}

// Pre-packed B fragment: 512 contiguous bf16 per (nt,kc); lane reads 16.
__device__ __forceinline__ v16bf load_b_frag(const __bf16* w, int lane, int nt, int kc) {
  return *(const v16bf*)(w + (((nt << 1) + kc) << 9) + (lane << 4));
}

// ---------------------------------------------------------------------------
// One recurrent step: h = tanh(t @ Wx + h @ Wh + b) on a 16x64 tile.
// t rows come from global memory (per-edge row pointer keyed by lane&15),
// h feedback round-trips through this wave's LDS tile (C-layout -> A-layout).
// ---------------------------------------------------------------------------
__device__ __forceinline__ void rsn_step(const float* tin_row, bool has_h, float* hl,
                                         const __bf16* wx, const __bf16* wh,
                                         const float* bvec, int lane) {
  int hi = (lane >> 4) & 1;
  int nc = lane & 15;
  v16bf at0 = load_a_frag(tin_row, lane, 0);
  v16bf at1 = load_a_frag(tin_row, lane, 1);
  v16bf ah0 = {}, ah1 = {};
  if (has_h) {
    const float* hrow = hl + (lane & 15) * HD;
    ah0 = load_a_frag(hrow, lane, 0);
    ah1 = load_a_frag(hrow, lane, 1);
  }
  __syncthreads();  // prior-phase LDS reads done before we overwrite the tile
#pragma unroll
  for (int nt = 0; nt < 4; ++nt) {
    v8f acc = {};
    acc = WMMA_BF16(at0, load_b_frag(wx, lane, nt, 0), acc);
    acc = WMMA_BF16(at1, load_b_frag(wx, lane, nt, 1), acc);
    if (has_h) {
      acc = WMMA_BF16(ah0, load_b_frag(wh, lane, nt, 0), acc);
      acc = WMMA_BF16(ah1, load_b_frag(wh, lane, nt, 1), acc);
    }
    float bv = bvec[nt * 16 + nc];
#pragma unroll
    for (int r = 0; r < 8; ++r)  // C layout: VGPR r -> M = r + hi*8, N = lane&15
      hl[(r + hi * 8) * HD + nt * 16 + nc] = TANHF(acc[r] + bv);
  }
  __syncthreads();
}

__device__ __forceinline__ void rsn_encode(const float* r0, const float* r1,
                                           const float* r2, float* hl,
                                           const __bf16* wx, const __bf16* wh,
                                           const float* bvec, int lane) {
  rsn_step(r0, false, hl, wx, wh, bvec, lane);
  rsn_step(r1, true, hl, wx, wh, bvec, lane);
  rsn_step(r2, true, hl, wx, wh, bvec, lane);
}

// lane -> (edge = lane>>1, head = lane&1); dot(h_inst[e,h,:], attn[h,:]) * sigmoid(eta[h])
__device__ __forceinline__ float head_score(const float* hl, const float* attn,
                                            const float* eta, int lane) {
  int ed = lane >> 1, hh = lane & 1;
  const float* rp = hl + ed * HD + hh * 32;
  const float* ap = attn + hh * 32;
  float s = 0.f;
#pragma unroll
  for (int d = 0; d < 32; ++d) s += rp[d] * ap[d];
  return s / (1.f + __expf(-eta[hh]));
}

// ---------------------------------------------------------------------------
// Kernel: pack the five 64x64 weights into WMMA B-fragment layout (bf16)
// wpack[w][(nt*2+kc)*512 + lane*16 + j] = W[w][(kofs(lane,j)+32*kc)*64 + nt*16 + (lane&15)]
// ---------------------------------------------------------------------------
__global__ void pack_weights(const float* Wl, const float* Wx0, const float* Wh0,
                             const float* Wx1, const float* Wh1, __bf16* wpack) {
  const float* Ws[5] = {Wl, Wx0, Wh0, Wx1, Wh1};
  for (int o = blockIdx.x * blockDim.x + threadIdx.x; o < 5 * 4096;
       o += blockDim.x * gridDim.x) {
    int w = o >> 12, rem = o & 4095;
    int f = rem >> 9, li = rem & 511;
    int nt = f >> 1, kc = f & 1;
    int l = li >> 4, j = li & 15;
    int K = kofs(l, j) + 32 * kc;
    int Nc = nt * 16 + (l & 15);
    wpack[o] = (__bf16)Ws[w][K * 64 + Nc];
  }
}

__global__ void init_ws(unsigned* smax, float* zeros, int nzero, int nsmax) {
  int i = blockIdx.x * blockDim.x + threadIdx.x;
  if (i < nzero) zeros[i] = 0.f;
  if (i < nsmax) smax[i] = 0x007FFFFFu;  // fenc(-inf)
}

// ---------------------------------------------------------------------------
// Kernel: h = x @ W_lin + b_lin  ([N,64]@[64,64], WMMA, one 16-row tile/wave)
// ---------------------------------------------------------------------------
__global__ __launch_bounds__(128) void node_linear(const float* __restrict__ x,
                                                   const float* __restrict__ b_lin,
                                                   const __bf16* __restrict__ wlin,
                                                   float* __restrict__ hout, int ntiles) {
  int wave = threadIdx.x >> 5, lane = threadIdx.x & 31;
  int tile = blockIdx.x * 4 + wave;
  if (tile >= ntiles) return;  // no LDS/sync in this kernel
  int hi = (lane >> 4) & 1, nc = lane & 15;
  const float* xrow = x + (size_t)(tile * 16 + nc) * HD;
  v16bf a0 = load_a_frag(xrow, lane, 0);
  v16bf a1 = load_a_frag(xrow, lane, 1);
#pragma unroll
  for (int nt = 0; nt < 4; ++nt) {
    v8f acc = {};
    acc = WMMA_BF16(a0, load_b_frag(wlin, lane, nt, 0), acc);
    acc = WMMA_BF16(a1, load_b_frag(wlin, lane, nt, 1), acc);
    float bv = b_lin[nt * 16 + nc];
#pragma unroll
    for (int r = 0; r < 8; ++r)
      hout[(size_t)(tile * 16 + r + hi * 8) * HD + nt * 16 + nc] = acc[r] + bv;
  }
}

// ---------------------------------------------------------------------------
// Edge pass 1: both encoders (80 WMMA / tile), edge scores, segment-max,
// prob-gated score sums. One 16-edge tile per wave, 4 waves/block.
// ---------------------------------------------------------------------------
__global__ __launch_bounds__(128) void edge_pass1(
    const float* __restrict__ h, const float* __restrict__ ef0,
    const float* __restrict__ ef1, const int* __restrict__ src,
    const int* __restrict__ dst, const int* __restrict__ prob,
    const __bf16* __restrict__ wpack, const float* __restrict__ b0,
    const float* __restrict__ b1, const float* __restrict__ attn0,
    const float* __restrict__ eta0, const float* __restrict__ attn1,
    const float* __restrict__ eta1, const float* __restrict__ delta,
    float* __restrict__ score_out, unsigned* __restrict__ smax,
    float* __restrict__ msum, float* __restrict__ mcnt,
    float* __restrict__ ssum, float* __restrict__ scnt) {
  __shared__ float lds[4][16 * HD];
  int wave = threadIdx.x >> 5, lane = threadIdx.x & 31;
  int tile = blockIdx.x * 4 + wave;  // grid divides E/16 exactly
  float* hl = lds[wave];
  int e = tile * 16 + (lane & 15);
  const float* hu_row = h + (size_t)src[e] * HD;
  const float* hv_row = h + (size_t)dst[e] * HD;
  const float* e0_row = ef0 + (size_t)e * HD;
  const float* e1_row = ef1 + (size_t)e * HD;
  // warm the HBM streams for recurrent step 2 while step 1 runs on WMMA
  __builtin_prefetch(e0_row, 0, 0);
  __builtin_prefetch(e1_row, 0, 0);
  const __bf16* wx0 = wpack + 1 * 4096;
  const __bf16* wh0 = wpack + 2 * 4096;
  const __bf16* wx1 = wpack + 3 * 4096;
  const __bf16* wh1 = wpack + 4 * 4096;

  rsn_encode(hu_row, e0_row, hv_row, hl, wx0, wh0, b0, lane);
  float e0s = head_score(hl, attn0, eta0, lane);
  __syncthreads();
  rsn_encode(hu_row, e1_row, hv_row, hl, wx1, wh1, b1, lane);
  float e1s = head_score(hl, attn1, eta1, lane);

  int ed = lane >> 1, hh = lane & 1;
  float g = 1.f / (1.f + __expf(-delta[hh]));
  float es = (1.f - g) * TANHF(e0s) + g * TANHF(e1s);
  int ee = tile * 16 + ed;
  int dn = dst[ee];
  score_out[(size_t)ee * 2 + hh] = es;
  atomicMax(smax + (size_t)dn * 2 + hh, fenc(es));
  float other = __shfl_xor(es, 1, 32);
  if (hh == 0) {
    float sc = 0.5f * (es + other);  // mean over heads
    int pv = prob[ee];
    if (pv > 1) {
      atomicAdd(msum + dn, sc); atomicAdd(mcnt + dn, 1.f);
    } else if (pv == 1) {
      atomicAdd(ssum + dn, sc); atomicAdd(scnt + dn, 1.f);
    }
  }
}

// ---------------------------------------------------------------------------
// Edge pass 2: recompute encoder-0 (40 WMMA / tile), ex = exp(s - smax[dst]),
// accumulate denom and UN-normalized  sum(hu*ex), sum(h_inst0*ex) (L2 atomics).
// ---------------------------------------------------------------------------
__global__ __launch_bounds__(128) void edge_pass2(
    const float* __restrict__ h, const float* __restrict__ ef0,
    const int* __restrict__ src, const int* __restrict__ dst,
    const __bf16* __restrict__ wpack, const float* __restrict__ b0,
    const float* __restrict__ score_in, const unsigned* __restrict__ smax,
    float* __restrict__ denom, float* __restrict__ ft, float* __restrict__ ft2) {
  __shared__ float lds[4][16 * HD];
  __shared__ float exs[4][32];
  __shared__ int dns[4][16];
  int wave = threadIdx.x >> 5, lane = threadIdx.x & 31;
  int tile = blockIdx.x * 4 + wave;
  float* hl = lds[wave];
  int e = tile * 16 + (lane & 15);
  const float* hu_row = h + (size_t)src[e] * HD;
  const float* hv_row = h + (size_t)dst[e] * HD;
  const float* e0_row = ef0 + (size_t)e * HD;
  __builtin_prefetch(e0_row, 0, 0);

  rsn_encode(hu_row, e0_row, hv_row, hl,
             wpack + 1 * 4096, wpack + 2 * 4096, b0, lane);

  int ed = lane >> 1, hh = lane & 1;
  int ee = tile * 16 + ed;
  int dn = dst[ee];
  float es = score_in[(size_t)ee * 2 + hh];
  float ex = __expf(es - fdec(smax[(size_t)dn * 2 + hh]));
  atomicAdd(denom + (size_t)dn * 2 + hh, ex);
  exs[wave][lane] = ex;
  if (hh == 0) dns[wave][ed] = dn;
  __syncthreads();
#pragma unroll 4
  for (int it = 0; it < 32; ++it) {  // 16 edges x 64 cols / 32 lanes
    int idx = it * 32 + lane;
    int ed2 = idx >> 6, col = idx & 63;
    int e2 = tile * 16 + ed2;
    int d2 = dns[wave][ed2];
    float w = exs[wave][ed2 * 2 + (col >> 5)];
    float huv = h[(size_t)src[e2] * HD + col];
    atomicAdd(ft + (size_t)d2 * HD + col, huv * w);
    atomicAdd(ft2 + (size_t)d2 * HD + col, hl[ed2 * HD + col] * w);
  }
}

// ---------------------------------------------------------------------------
// Finalize: rst = (ft/denom)@Wsrc + bsrc + (ft2/denom)@Wedge + bedge + h, relu;
// plus prob-gated segment means. One thread per (n, col).
// ---------------------------------------------------------------------------
__global__ __launch_bounds__(256) void finalize(
    const float* __restrict__ hbuf, const float* __restrict__ ft,
    const float* __restrict__ ft2, const float* __restrict__ denom,
    const float* __restrict__ Wsrc, const float* __restrict__ bsrc,
    const float* __restrict__ Wedge, const float* __restrict__ bedge,
    const float* __restrict__ msum, const float* __restrict__ mcnt,
    const float* __restrict__ ssum, const float* __restrict__ scnt,
    float* __restrict__ out) {
  int idx = blockIdx.x * blockDim.x + threadIdx.x;
  if (idx >= NN * HD) return;
  int n = idx >> 6, c = idx & 63, hh = c >> 5, dN = c & 31;
  float dd = fmaxf(denom[(size_t)n * 2 + hh], 1e-9f);
  float inv = 1.f / dd;
  const float* f1 = ft + (size_t)n * HD + hh * 32;
  const float* f2 = ft2 + (size_t)n * HD + hh * 32;
  float acc = 0.f;
#pragma unroll 8
  for (int d = 0; d < 32; ++d)
    acc += f1[d] * Wsrc[d * 32 + dN] + f2[d] * Wedge[d * 32 + dN];
  acc = acc * inv + bsrc[dN] + bedge[dN] + hbuf[(size_t)n * HD + c];
  out[(size_t)n * HD + c] = fmaxf(acc, 0.f);
  if (c == 0) {
    float mc = mcnt[n], sc = scnt[n];
    float mm = msum[n] / (mc > 0.f ? mc : 1.f);
    float ms = ssum[n] / (sc > 0.f ? sc : 1.f);
    float ind = (mc > 0.f && sc > 0.f) ? 1.f : 0.f;
    out[(size_t)NN * HD + n] = mm * ind;
    out[(size_t)NN * HD + NN + n] = ms * ind;
  }
}

// ---------------------------------------------------------------------------
extern "C" void kernel_launch(void* const* d_in, const int* in_sizes, int n_in,
                              void* d_out, int out_size, void* d_ws, size_t ws_size,
                              hipStream_t stream) {
  const float* x     = (const float*)d_in[0];
  const float* ef0   = (const float*)d_in[1];
  const float* ef1   = (const float*)d_in[2];
  const int*   src   = (const int*)d_in[3];
  const int*   dst   = (const int*)d_in[4];
  const int*   prob  = (const int*)d_in[5];
  const float* W_lin = (const float*)d_in[6];
  const float* b_lin = (const float*)d_in[7];
  const float* Wx0   = (const float*)d_in[8];
  const float* Wh0   = (const float*)d_in[9];
  const float* b0    = (const float*)d_in[10];
  const float* Wx1   = (const float*)d_in[11];
  const float* Wh1   = (const float*)d_in[12];
  const float* b1    = (const float*)d_in[13];
  const float* attn0 = (const float*)d_in[14];
  const float* eta0  = (const float*)d_in[15];
  const float* attn1 = (const float*)d_in[16];
  const float* eta1  = (const float*)d_in[17];
  const float* delta = (const float*)d_in[18];
  const float* Wsrc  = (const float*)d_in[19];
  const float* bsrc  = (const float*)d_in[20];
  const float* Wedge = (const float*)d_in[21];
  const float* bedge = (const float*)d_in[22];

  // workspace carve-up (~46.6 MB)
  char* p = (char*)d_ws;
  __bf16*   wpack = (__bf16*)p;   p += (size_t)5 * 4096 * sizeof(__bf16);
  float*    hbuf  = (float*)p;    p += (size_t)NN * HD * 4;
  float*    score = (float*)p;    p += (size_t)EE * 2 * 4;
  unsigned* smax  = (unsigned*)p; p += (size_t)NN * 2 * 4;
  float*    denom = (float*)p;    p += (size_t)NN * 2 * 4;  // zero region starts here
  float*    ft    = (float*)p;    p += (size_t)NN * HD * 4;
  float*    ft2   = (float*)p;    p += (size_t)NN * HD * 4;
  float*    msum  = (float*)p;    p += (size_t)NN * 4;
  float*    mcnt  = (float*)p;    p += (size_t)NN * 4;
  float*    ssum  = (float*)p;    p += (size_t)NN * 4;
  float*    scnt  = (float*)p;    p += (size_t)NN * 4;

  pack_weights<<<20, 256, 0, stream>>>(W_lin, Wx0, Wh0, Wx1, Wh1, wpack);
  int nzero = NN * (2 + 2 * HD + 4);
  init_ws<<<(nzero + 255) / 256, 256, 0, stream>>>(smax, denom, nzero, NN * 2);
  node_linear<<<(3125 + 3) / 4, 128, 0, stream>>>(x, b_lin, wpack, hbuf, 3125);
  edge_pass1<<<EE / 64, 128, 0, stream>>>(hbuf, ef0, ef1, src, dst, prob, wpack,
                                          b0, b1, attn0, eta0, attn1, eta1, delta,
                                          score, smax, msum, mcnt, ssum, scnt);
  edge_pass2<<<EE / 64, 128, 0, stream>>>(hbuf, ef0, src, dst, wpack, b0, score,
                                          smax, denom, ft, ft2);
  finalize<<<(NN * HD + 255) / 256, 256, 0, stream>>>(hbuf, ft, ft2, denom, Wsrc,
                                                      bsrc, Wedge, bedge, msum,
                                                      mcnt, ssum, scnt, (float*)d_out);
}
